// units2indices_49134425866734
// MI455X (gfx1250) — matrix operand order repaired and checked by reading, not compiled, not run on hardware
//
#include <hip/hip_runtime.h>

typedef __attribute__((ext_vector_type(16))) _Float16 v16h;
typedef __attribute__((ext_vector_type(8)))  float    v8f;

// bipolar {-1,+1} -> bit {0,1} as f16 (sign bit clear => +1 => bit 1).
// Lowers to v_cmp + v_cndmask_b16 (2 VALU, 16-bit write).
__device__ __forceinline__ _Float16 bit_of(float f) {
    return (__float_as_int(f) >= 0) ? (_Float16)1.0f : (_Float16)0.0f;
}

// WMMA path, num_bits == 10, FULL TILES ONLY (no bounds checks anywhere).
// Tile = 16x16 = 256 groups = 2560 input floats. A row r holds the 160 bits of
// groups [tile*256+16r, +16) as f16; B[k][c] = 2^(9-(k-10c)) for k in [10c,10c+10),
// else 0. C[r][c] accumulated over 5 K=32 WMMAs = exact group index (<=1023).
__global__ __launch_bounds__(256) void units2indices_wmma10(
        const float* __restrict__ in, int* __restrict__ out, int nFullTiles) {
    const int lane = threadIdx.x & 31;
    const int wave = threadIdx.x >> 5;
    const int hi   = lane >> 4;     // 0: lanes 0-15, 1: lanes 16-31
    const int c    = lane & 15;     // B/C column; also the A row this lane holds

    // Constant B fragments (weights), lane-resident, computed once.
    // B k order chosen here (contiguous): lanes 0-15 -> K = i, lanes 16-31 -> K = 16 + i.
    v16h b[5];
#pragma unroll
    for (int p = 0; p < 5; ++p) {
#pragma unroll
        for (int i = 0; i < 16; ++i) {
            int k = 32 * p + 16 * hi + i;   // K within the 160-bit row
            int j = k - 10 * c;             // bit position inside group c
            float w = (j >= 0 && j < 10) ? (float)(1 << (9 - j)) : 0.0f;
            b[p][i] = (_Float16)w;
        }
    }

    const long long tileStride = (long long)gridDim.x * 8;
    // Per-lane fixed offset within a tile: row base + K-half offset.
    const long long laneOff = 160LL * c + 16LL * hi;

    for (long long tile = (long long)blockIdx.x * 8 + wave; tile < nFullTiles;
         tile += tileStride) {
        const float* rowPtr = in + tile * 2560 + laneOff;

        // Hint next tile's lines into cache while the WMMAs below execute.
        __builtin_prefetch(rowPtr + tileStride * 2560, 0, 3);

        v8f acc = {};
#pragma unroll
        for (int p = 0; p < 5; ++p) {
            // A k order chosen here (contiguous): lanes 0-15: elems 0-15 -> K 0-15;
            // lanes 16-31: elems 0-15 -> K 16-31 (matches B).
            // Each 8-elem chunk = 8 consecutive floats, 32B aligned ->
            // two global_load_b128; every 128B line is fully consumed.
            const float4* q0 = (const float4*)(rowPtr + 32 * p);      // elems 0-7
            const float4* q1 = (const float4*)(rowPtr + 32 * p + 8);  // elems 8-15
            float4 f0 = q0[0], f1 = q0[1], f2 = q1[0], f3 = q1[1];
            v16h a;
            a[0]  = bit_of(f0.x); a[1]  = bit_of(f0.y);
            a[2]  = bit_of(f0.z); a[3]  = bit_of(f0.w);
            a[4]  = bit_of(f1.x); a[5]  = bit_of(f1.y);
            a[6]  = bit_of(f1.z); a[7]  = bit_of(f1.w);
            a[8]  = bit_of(f2.x); a[9]  = bit_of(f2.y);
            a[10] = bit_of(f2.z); a[11] = bit_of(f2.w);
            a[12] = bit_of(f3.x); a[13] = bit_of(f3.y);
            a[14] = bit_of(f3.z); a[15] = bit_of(f3.w);
            // (neg_a, A, neg_b, B, c_mod, C, reuse_a, reuse_b)
            acc = __builtin_amdgcn_wmma_f32_16x16x32_f16(
                false, a, false, b[p], (short)0, acc, false, false);
        }

        // C layout: VGPR v -> M = v + 8*hi, N = c. Full tile => always in range.
        int* outTile = out + tile * 256 + 16LL * 8 * hi + c;
#pragma unroll
        for (int v = 0; v < 8; ++v) {
            outTile[16 * v] = (int)acc[v];
        }
    }
}

// Generic scalar kernel: covers groups [firstGroup, nGroups) for the tile tail,
// and the whole range when num_bits != 10. MSB-first pack, bounds-checked.
__global__ void units2indices_generic(const float* __restrict__ in,
                                      int* __restrict__ out,
                                      long long nFloats, int nb,
                                      long long firstGroup, long long nGroups) {
    long long g = firstGroup + (long long)blockIdx.x * blockDim.x + threadIdx.x;
    if (g >= nGroups) return;
    long long base = g * nb;
    int idx = 0;
    for (int j = 0; j < nb; ++j) {
        long long p = base + j;
        unsigned u = (p < nFloats) ? __float_as_uint(in[p]) : 0x80000000u;
        idx = (idx << 1) | (int)((~u >> 31) & 1u);
    }
    out[g] = idx;
}

extern "C" void kernel_launch(void* const* d_in, const int* in_sizes, int n_in,
                              void* d_out, int out_size, void* d_ws, size_t ws_size,
                              hipStream_t stream) {
    const float* in = (const float*)d_in[0];
    int* out = (int*)d_out;
    long long nFloats = (long long)in_sizes[0];
    long long nGroups = (long long)out_size;
    // num_bits lives in device memory (d_in[1]); recover it deterministically
    // on the host from the shapes: nFloats / nGroups.
    int nb = (nGroups > 0) ? (int)(nFloats / nGroups) : 10;

    if (nb == 10) {
        // A tile is full iff all 2560 floats and all 256 groups are in range.
        long long byGroups = nGroups / 256;
        long long byFloats = nFloats / 2560;
        long long nFullTiles = byGroups < byFloats ? byGroups : byFloats;

        if (nFullTiles > 0) {
            long long blocks = (nFullTiles + 7) / 8;  // 8 waves/block, 1 tile/wave/iter
            if (blocks > 2048) blocks = 2048;
            units2indices_wmma10<<<(int)blocks, 256, 0, stream>>>(
                in, out, (int)nFullTiles);
        }
        long long firstTail = nFullTiles * 256;
        long long tailGroups = nGroups - firstTail;
        if (tailGroups > 0) {
            int blocks = (int)((tailGroups + 255) / 256);
            units2indices_generic<<<blocks, 256, 0, stream>>>(
                in, out, nFloats, nb, firstTail, nGroups);
        }
    } else {
        if (nGroups > 0) {
            int blocks = (int)((nGroups + 255) / 256);
            units2indices_generic<<<blocks, 256, 0, stream>>>(
                in, out, nFloats, nb, 0, nGroups);
        }
    }
}
